// conv_slim_capsules_2619930050938
// MI455X (gfx1250) — matrix operand order repaired and checked by reading, not compiled, not run on hardware
//
#include <hip/hip_runtime.h>

typedef __attribute__((ext_vector_type(16))) _Float16 v16h;
typedef __attribute__((ext_vector_type(8)))  _Float16 v8h;
typedef __attribute__((ext_vector_type(8)))  float    v8f;

namespace {
constexpr int kBS = 16, kCI = 32, kNI = 8, kHI = 20, kWI = 20;
constexpr int kCO = 32, kNO = 8;
constexpr int kNOUT = kCO * kNO;        // 256 = CO*NO
constexpr int kKDIM = kNI * 9;          // 72  = NI*3*3
constexpr int kKPAD = 96;               // pad K to 3 x 32 for WMMA 16x16x32
constexpr int kITERS = 3;
constexpr int kPIX  = kHI * kWI;        // 400

// Dynamic-LDS carve (bytes). Total 153,600 B (<320 KB/WG per CDNA5 ISA; 2 WG/WGP).
constexpr size_t OFF_AH   = 0;                                     // 32*96*2 = 6144
constexpr size_t OFF_AL   = OFF_AH   + (size_t)kCI  * kKPAD * 2;
constexpr size_t OFF_BH   = OFF_AL   + (size_t)kCI  * kKPAD * 2;   // 256*96*2 = 49152
constexpr size_t OFF_BL   = OFF_BH   + (size_t)kNOUT* kKPAD * 2;
constexpr size_t OFF_VOTE = OFF_BL   + (size_t)kNOUT* kKPAD * 2;   // 32*256*4 = 32768
constexpr size_t OFF_LOG  = OFF_VOTE + (size_t)kCI  * kNOUT * 4;   // 32*32*4
constexpr size_t OFF_RTE  = OFF_LOG  + (size_t)kCI  * kCO   * 4;
constexpr size_t OFF_PRE  = OFF_RTE  + (size_t)kCI  * kCO   * 4;   // 256*4
constexpr size_t OFF_ACT  = OFF_PRE  + (size_t)kNOUT * 4;
constexpr size_t SMEM_BYTES = OFF_ACT + (size_t)kNOUT * 4;         // 153600
} // namespace

// One workgroup per (batch, h, w) output position. Fully fused:
// im2col patch -> hi/lo f16 split -> WMMA 16x16x32 f32-accumulate GEMM
// (votes stay in LDS) -> 3 dynamic-routing iterations in LDS -> activation out.
__global__ __launch_bounds__(256, 1) void caps_routing_kernel(
    const float* __restrict__ x, const float* __restrict__ Wg,
    const float* __restrict__ bias, float* __restrict__ out)
{
  extern __shared__ char smem[];
  _Float16* Ah    = (_Float16*)(smem + OFF_AH);
  _Float16* Al    = (_Float16*)(smem + OFF_AL);
  _Float16* Bh    = (_Float16*)(smem + OFF_BH);
  _Float16* Bl    = (_Float16*)(smem + OFF_BL);
  float*    votes = (float*)   (smem + OFF_VOTE);   // [ci][co*no]
  float*    logit = (float*)   (smem + OFF_LOG);    // [ci][co]
  float*    route = (float*)   (smem + OFF_RTE);    // [ci][co]
  float*    prea  = (float*)   (smem + OFF_PRE);    // [co*no]
  float*    act   = (float*)   (smem + OFF_ACT);    // [co*no]

  const int tid = threadIdx.x;
  const int b   = blockIdx.x / kPIX;
  const int pix = blockIdx.x % kPIX;
  const int h   = pix / kWI;
  const int w   = pix % kWI;

  // ---- Stage B = weights [n][k] (W's native layout), hi/lo f16 split.
  // Coalesced float4 loads: lanes read contiguous 16B chunks (global_load_b128).
  // 72 floats/row = 18 float4 -> no float4 straddles a row.
  {
    const float4* W4 = (const float4*)Wg;                 // 18432 f32 = 4608 f4
    for (int i = tid; i < (kNOUT * kKDIM) / 4; i += 256) {
      const float4 v4 = W4[i];
      const int base = i * 4;
      const int n  = base / kKDIM;
      const int k0 = base - n * kKDIM;
      const float vs[4] = {v4.x, v4.y, v4.z, v4.w};
#pragma unroll
      for (int j = 0; j < 4; ++j) {
        const float v = vs[j];
        const _Float16 hv = (_Float16)v;
        Bh[n * kKPAD + k0 + j] = hv;
        Bl[n * kKPAD + k0 + j] = (_Float16)(v - (float)hv);
      }
    }
    // zero the K padding (k = 72..95) for all 256 rows
    for (int i = tid; i < kNOUT * (kKPAD - kKDIM); i += 256) {
      const int n = i / (kKPAD - kKDIM);
      const int k = kKDIM + (i - n * (kKPAD - kKDIM));
      Bh[n * kKPAD + k] = (_Float16)0.f;
      Bl[n * kKPAD + k] = (_Float16)0.f;
    }
  }
  // ---- Stage A = im2col patch [ci][k], k = ni*9 + kh*3 + kw, zero border pad.
  for (int idx = tid; idx < kCI * kKPAD; idx += 256) {
    const int ci = idx / kKPAD;
    const int k  = idx % kKPAD;
    float v = 0.f;
    if (k < kKDIM) {
      const int ni = k / 9, r = k - ni * 9;
      const int kh = r / 3, kw = r - kh * 3;
      const int gh = h + kh - 1, gw = w + kw - 1;
      if (gh >= 0 && gh < kHI && gw >= 0 && gw < kWI)
        v = x[((((size_t)b * kCI + ci) * kNI + ni) * kHI + gh) * kWI + gw];
    }
    const _Float16 hv = (_Float16)v;
    Ah[idx] = hv;
    Al[idx] = (_Float16)(v - (float)hv);
  }
  for (int i = tid; i < kCI * kCO; i += 256) logit[i] = 0.f;
  __syncthreads();

  // ---- WMMA GEMM: votes[32 x 256] = A[32 x 96] * B^T[96 x 256]
  const int wave   = tid >> 5;
  const int lane   = tid & 31;
  const int lrow   = lane & 15;
  const int khalf  = lane >> 4;
  const int mtile  = wave & 1;               // 2 M-tiles of 16 (ci)
  const int ntbase = (wave >> 1) * 4;        // 4 groups x 4 N-tiles of 16

  // A fragments per ISA 16-bit 16x32 layout:
  // lane(khalf=0): K = {0..7, 16..23}; lane(khalf=1): K = {8..15, 24..31}.
  v16h aH[3], aL[3];
  const int am = mtile * 16 + lrow;
#pragma unroll
  for (int kt = 0; kt < 3; ++kt) {
    const _Float16* ph = &Ah[am * kKPAD + kt * 32 + khalf * 8];
    v8h h0 = *(const v8h*)(ph);
    v8h h1 = *(const v8h*)(ph + 16);
    const _Float16* pl = &Al[am * kKPAD + kt * 32 + khalf * 8];
    v8h l0 = *(const v8h*)(pl);
    v8h l1 = *(const v8h*)(pl + 16);
#pragma unroll
    for (int i = 0; i < 8; ++i) {
      aH[kt][i] = h0[i]; aH[kt][i + 8] = h1[i];
      aL[kt][i] = l0[i]; aL[kt][i + 8] = l1[i];
    }
  }

  // K-outer / N-tile-inner: 4 independent accumulator chains for WMMA-pipe ILP.
  v8f acc[4] = {};
#pragma unroll
  for (int kt = 0; kt < 3; ++kt) {
#pragma unroll
    for (int j = 0; j < 4; ++j) {
      const int bn = (ntbase + j) * 16 + lrow;
      // B fragment matches A: lane(khalf) holds K = khalf*8 + {0..7, 16..23}.
      const _Float16* pb = &Bh[bn * kKPAD + kt * 32 + khalf * 8];
      v8h b0 = *(const v8h*)(pb);
      v8h b1 = *(const v8h*)(pb + 16);
      const _Float16* ql = &Bl[bn * kKPAD + kt * 32 + khalf * 8];
      v8h c0 = *(const v8h*)(ql);
      v8h c1 = *(const v8h*)(ql + 16);
      v16h bH, bL;
#pragma unroll
      for (int i = 0; i < 8; ++i) {
        bH[i] = b0[i]; bH[i + 8] = b1[i];
        bL[i] = c0[i]; bL[i + 8] = c1[i];
      }
      // fp32-accurate product via hi/lo split: AhBh + AhBl + AlBh (AlBl ~ 2^-22)
      acc[j] = __builtin_amdgcn_wmma_f32_16x16x32_f16(false, aH[kt], false, bH, (short)0, acc[j], false, false);
      acc[j] = __builtin_amdgcn_wmma_f32_16x16x32_f16(false, aH[kt], false, bL, (short)0, acc[j], false, false);
      acc[j] = __builtin_amdgcn_wmma_f32_16x16x32_f16(false, aL[kt], false, bH, (short)0, acc[j], false, false);
    }
  }
  // C layout: VGPR r, lanes 0-15 -> M=r, lanes 16-31 -> M=r+8; N = lane&15.
#pragma unroll
  for (int j = 0; j < 4; ++j) {
#pragma unroll
    for (int r = 0; r < 8; ++r)
      votes[(mtile * 16 + khalf * 8 + r) * kNOUT + (ntbase + j) * 16 + lrow] = acc[j][r];
  }
  __syncthreads();

  // ---- Dynamic routing, entirely in LDS (per-position independent).
  for (int it = 0; it < kITERS; ++it) {
    // softmax over CO per ci row (32 rows, one thread each)
    if (tid < kCI) {
      const int ci = tid;
      float mx = -1e30f;
      for (int co = 0; co < kCO; ++co) mx = fmaxf(mx, logit[ci * kCO + co]);
      float s = 0.f;
      for (int co = 0; co < kCO; ++co) {
        float e = __expf(logit[ci * kCO + co] - mx);
        route[ci * kCO + co] = e;
        s += e;
      }
      const float inv = 1.f / s;
      for (int co = 0; co < kCO; ++co) route[ci * kCO + co] *= inv;
    }
    __syncthreads();

    // preact[n] = sum_ci route[ci][co] * votes[ci][n] + bias[n]   (n = co*8+no)
    {
      const int n = tid;
      const int co = n >> 3;
      float p = bias[n];
      for (int ci = 0; ci < kCI; ++ci)
        p = fmaf(route[ci * kCO + co], votes[ci * kNOUT + n], p);
      prea[n] = p;
      __syncthreads();
      // squash over the 8 atoms of this co
      float n2 = 0.f;
#pragma unroll
      for (int q = 0; q < kNO; ++q) {
        float t = prea[co * kNO + q];
        n2 = fmaf(t, t, n2);
      }
      act[n] = p * (n2 / ((1.f + n2) * sqrtf(n2 + 1e-9f)));
    }
    __syncthreads();

    if (it < kITERS - 1) {
      // logits[ci][co] += sum_no votes[ci][co*8+no] * act[co*8+no]
      for (int p = tid; p < kCI * kCO; p += 256) {
        const int ci = p >> 5;
        const int co = p & 31;
        float d = 0.f;
#pragma unroll
        for (int q = 0; q < kNO; ++q)
          d = fmaf(votes[ci * kNOUT + co * kNO + q], act[co * kNO + q], d);
        logit[p] += d;
      }
      __syncthreads();
    }
  }

  // ---- activation -> out, layout (bs, co, no, ho, wo)
  out[((size_t)b * kNOUT + tid) * kPIX + pix] = act[tid];
}

extern "C" void kernel_launch(void* const* d_in, const int* in_sizes, int n_in,
                              void* d_out, int out_size, void* d_ws, size_t ws_size,
                              hipStream_t stream) {
  (void)in_sizes; (void)n_in; (void)out_size; (void)d_ws; (void)ws_size;
  const float* x    = (const float*)d_in[0];
  const float* Wg   = (const float*)d_in[1];
  const float* bias = (const float*)d_in[2];
  float* out        = (float*)d_out;

  dim3 grid(kBS * kPIX);   // 6400 workgroups, one per (b, h, w)
  dim3 block(256);         // 8 wave32 waves
  caps_routing_kernel<<<grid, block, SMEM_BYTES, stream>>>(x, Wg, bias, out);
}
